// Agent_LSTM_PPO_61856118997661
// MI455X (gfx1250) — matrix operand order repaired and checked by reading, not compiled, not run on hardware
//
#include <hip/hip_runtime.h>
#include <hip/hip_bf16.h>
#include <math.h>

// ---------------- problem constants ----------------
#define OBS_DIMc 128
#define HIDc     256
#define SEQc     512
#define BATCHc   256
#define COMBc    384          // OBS_DIM + HID
#define NGATEc   1024         // 4 * HID
#define KTILES   12           // COMB / 32
#define NTILES   64           // NGATE / 16

typedef __attribute__((ext_vector_type(16))) _Float16 v16h;
typedef __attribute__((ext_vector_type(8)))  float    v8f;

// workspace layout (bytes)
#define OBS16_BYTES ((size_t)BATCHc * SEQc * OBS_DIMc * 2)        // 33,554,432
#define WPACK_BYTES ((size_t)KTILES * NTILES * 32 * 16 * 2)       // 786,432
#define BIAS_BYTES  ((size_t)NGATEc * 4)                          // 4,096
#define LDS_BYTES   (16*COMBc*2 + 16*NGATEc*4 + 16*HIDc*4)        // 94,208

// ---------------- prep: obs f32 -> f16 ----------------
struct __attribute__((aligned(8))) H4 { _Float16 x, y, z, w; };

__global__ void k_convert_obs(const float* __restrict__ src, _Float16* __restrict__ dst) {
  size_t i = ((size_t)blockIdx.x * blockDim.x + threadIdx.x) * 4;
  float4 v = *(const float4*)(src + i);
  H4 o; o.x = (_Float16)v.x; o.y = (_Float16)v.y; o.z = (_Float16)v.z; o.w = (_Float16)v.w;
  *(H4*)(dst + i) = o;
}

// ---------------- prep: pack gate weights into WMMA B-fragment layout ----------------
// B-matrix 32x16 f16 fragment, contiguous k order (same as A): lanes 0-15 hold K=0..15 (2 per VGPR),
// lanes 16-31 hold K=16..31. Fragment storage: [kt][nt][lane][16 halves] contiguous.
__global__ void k_pack_w(const float* __restrict__ Wf, const float* __restrict__ Wi,
                         const float* __restrict__ Wc, const float* __restrict__ Wo,
                         _Float16* __restrict__ wpack) {
  int tid  = blockIdx.x * blockDim.x + threadIdx.x;   // 0 .. 393215
  int h    = tid & 15;
  int lane = (tid >> 4) & 31;
  int frag = tid >> 9;
  int kt = frag >> 6, nt = frag & 63;
  int K = kt * 32 + ((lane >> 4) << 4) + h;           // 0..383
  int N = nt * 16 + (lane & 15);                      // 0..1023
  int gate = N >> 8, col = N & 255;
  const float* W = (gate == 0) ? Wf : (gate == 1) ? Wi : (gate == 2) ? Wc : Wo;
  wpack[tid] = (_Float16)W[K * HIDc + col];
}

__global__ void k_pack_bias(const float* __restrict__ bf, const float* __restrict__ bi,
                            const float* __restrict__ bc, const float* __restrict__ bo,
                            float* __restrict__ biasg) {
  int tid = blockIdx.x * blockDim.x + threadIdx.x;    // 0..1023
  int gate = tid >> 8, j = tid & 255;
  const float* s = (gate == 0) ? bf : (gate == 1) ? bi : (gate == 2) ? bc : bo;
  biasg[tid] = s[j];
}

// ---------------- persistent LSTM kernel ----------------
// 16 blocks x 512 threads (16 waves). Block owns 16 batch rows for all 512 steps.
// Wave w owns gate columns [w*64, w*64+64) = 4 WMMA n-tiles (single gate region).
// Weights are streamed from L2 every step (768KB/block/step, L2-resident); the
// asm pointer-clobber below stops LICM from hoisting+spilling them to scratch.
__global__ void __launch_bounds__(512, 1)
k_lstm(const _Float16* __restrict__ obs16,
       const _Float16* __restrict__ wpack,
       const float*    __restrict__ biasg,
       float*          __restrict__ hlast) {
  extern __shared__ char smem[];
  _Float16* combA  = (_Float16*)smem;                        // [16][384] f16 (A operand rows)
  float*    gatesL = (float*)(smem + 16 * COMBc * 2);        // [16][1024] f32
  float*    cst    = (float*)(smem + 16 * COMBc * 2 + 16 * NGATEc * 4); // [16][256] f32

  const int tid  = threadIdx.x;
  const int lane = tid & 31;
  const int wv   = tid >> 5;                 // 0..15
  const int b0   = blockIdx.x * 16;          // batch tile origin

  // init: h-part of comb and cell state to zero (reference h0=c0=0)
  for (int i = tid; i < 16 * HIDc; i += 512) {
    int m = i >> 8, j = i & 255;
    combA[m * COMBc + OBS_DIMc + j] = (_Float16)0.0f;
    cst[i] = 0.0f;
  }

  // wave-uniform constants
  const int  mBase  = (lane >> 4) << 3;      // C/D row base: 0 or 8
  const int  nt0    = wv * 4;
  const int  gateid = wv >> 2;               // 0:f 1:i 2:c~ 3:o
  const bool isTanh = (gateid == 2);
  float bias[4];
  #pragma unroll
  for (int s = 0; s < 4; ++s) bias[s] = biasg[(nt0 + s) * 16 + (lane & 15)];

  union AF { v16h v; unsigned u[8]; };
  const unsigned* lds32 = (const unsigned*)combA;
  const int mA = lane & 15;
  const int kb = (lane >> 4) << 3;           // A kbase: 0 or 8

  for (int t = 0; t < SEQc; ++t) {
    // Clobber the weight base pointer every iteration: loads through it are now
    // formally loop-variant, so LLVM keeps them as in-loop global_load_b128
    // (L2 streaming) instead of hoisting 384 VGPRs of weights and spilling.
    const v16h* wfrag = (const v16h*)wpack;
    asm volatile("" : "+s"(wfrag));

    // (1) stage x_t (16 x 128 f16) into comb: first 256 threads, 16B each
    if (tid < 256) {
      int m  = tid >> 4;
      int k0 = (tid & 15) << 3;
      const _Float16* srcp = obs16 + ((size_t)(b0 + m) * SEQc + t) * OBS_DIMc + k0;
      const uint4 v = *(const uint4*)srcp;
      *(uint4*)(combA + m * COMBc + k0) = v;
      if (t + 1 < SEQc)                       // pull x_{t+1} toward the WGP early
        __builtin_prefetch(srcp + OBS_DIMc, 0, 1);
    }
    __syncthreads();   // x_t and h_{t-1} visible to all waves

    // (2) gate GEMM: 16x1024 = 16x384 @ 384x1024 (this wave: 4 n-tiles)
    AF a[KTILES];
    #pragma unroll
    for (int kt = 0; kt < KTILES; ++kt) {
      #pragma unroll
      for (int p = 0; p < 8; ++p) {           // A 16x32 f16, contiguous k = 16*(lane/16) + s (same as B pack)
        int h2 = p << 1;
        int K  = kt * 32 + (kb << 1) + h2;
        a[kt].u[p] = lds32[(mA * COMBc + K) >> 1];
      }
    }
    v8f acc0, acc1, acc2, acc3;
    #pragma unroll
    for (int r = 0; r < 8; ++r) { acc0[r] = bias[0]; acc1[r] = bias[1]; acc2[r] = bias[2]; acc3[r] = bias[3]; }
    #pragma unroll
    for (int kt = 0; kt < KTILES; ++kt) {
      v16h bf0 = wfrag[(size_t)(kt * NTILES + nt0 + 0) * 32 + lane];
      v16h bf1 = wfrag[(size_t)(kt * NTILES + nt0 + 1) * 32 + lane];
      v16h bf2 = wfrag[(size_t)(kt * NTILES + nt0 + 2) * 32 + lane];
      v16h bf3 = wfrag[(size_t)(kt * NTILES + nt0 + 3) * 32 + lane];
      acc0 = __builtin_amdgcn_wmma_f32_16x16x32_f16(false, a[kt].v, false, bf0, (short)0, acc0, false, false);
      acc1 = __builtin_amdgcn_wmma_f32_16x16x32_f16(false, a[kt].v, false, bf1, (short)0, acc1, false, false);
      acc2 = __builtin_amdgcn_wmma_f32_16x16x32_f16(false, a[kt].v, false, bf2, (short)0, acc2, false, false);
      acc3 = __builtin_amdgcn_wmma_f32_16x16x32_f16(false, a[kt].v, false, bf3, (short)0, acc3, false, false);
    }
    // activation (wave-uniform) + stage to gates LDS
    #pragma unroll
    for (int s = 0; s < 4; ++s) {
      v8f acc = (s == 0) ? acc0 : (s == 1) ? acc1 : (s == 2) ? acc2 : acc3;
      int n = (nt0 + s) * 16 + (lane & 15);
      #pragma unroll
      for (int r = 0; r < 8; ++r) {
        float x = acc[r];
        float y = isTanh ? tanhf(x) : (1.0f / (1.0f + __expf(-x)));
        gatesL[(mBase + r) * NGATEc + n] = y;
      }
    }
    __syncthreads();   // gates complete

    // (3) elementwise cell update: 4096 elems, 8 per thread
    #pragma unroll
    for (int i = 0; i < 8; ++i) {
      int e = tid + i * 512;
      int m = e >> 8, j = e & 255;
      float fg = gatesL[m * NGATEc + j];
      float ig = gatesL[m * NGATEc + HIDc + j];
      float cg = gatesL[m * NGATEc + 2 * HIDc + j];
      float og = gatesL[m * NGATEc + 3 * HIDc + j];
      float cn = fg * cst[e] + ig * cg;
      float hn = og * tanhf(cn);
      cst[e] = cn;
      combA[m * COMBc + OBS_DIMc + j] = (_Float16)hn;   // h_t feeds next step's A operand
      if (t == SEQc - 1) hlast[(size_t)(b0 + m) * HIDc + j] = hn;
    }
    // next iteration's step-(1) barrier orders these writes vs. the next GEMM
  }
}

// ---------------- head kernel: one block per batch row ----------------
__global__ void __launch_bounds__(256)
k_heads(const float* __restrict__ hlast, const int* __restrict__ action,
        const float* __restrict__ W1, const float* __restrict__ b1,
        const float* __restrict__ W2, const float* __restrict__ b2,
        const float* __restrict__ A1, const float* __restrict__ a1,
        const float* __restrict__ A2, const float* __restrict__ a2,
        const float* __restrict__ A3, const float* __restrict__ a3,
        const float* __restrict__ C1, const float* __restrict__ c1,
        const float* __restrict__ C2, const float* __restrict__ c2,
        const float* __restrict__ C3, const float* __restrict__ c3,
        float* __restrict__ out) {
  __shared__ float sh[HIDc];
  __shared__ float z1s[512];
  __shared__ float z2s[512];
  __shared__ float hA[64], hB[64], hA2[64], hB2[64];
  __shared__ float lg[32];
  __shared__ float valS;
  const int b = blockIdx.x, tid = threadIdx.x;

  sh[tid] = hlast[(size_t)b * HIDc + tid];
  __syncthreads();

  for (int j = tid; j < 512; j += 256) {       // z1 = relu(h @ W1 + b1)
    float s = b1[j];
    for (int k = 0; k < HIDc; ++k) s += sh[k] * W1[k * 512 + j];
    z1s[j] = fmaxf(s, 0.0f);
  }
  __syncthreads();
  for (int j = tid; j < 512; j += 256) {       // z2 = relu(z1 @ W2 + b2)
    float s = b2[j];
    for (int k = 0; k < 512; ++k) s += z1s[k] * W2[k * 512 + j];
    z2s[j] = fmaxf(s, 0.0f);
  }
  __syncthreads();
  if (tid < 64) {                              // actor layer 1
    float s = a1[tid];
    for (int k = 0; k < 512; ++k) s += z2s[k] * A1[k * 64 + tid];
    hA[tid] = tanhf(s);
  } else if (tid < 128) {                      // critic layer 1
    int j = tid - 64; float s = c1[j];
    for (int k = 0; k < 512; ++k) s += z2s[k] * C1[k * 64 + j];
    hB[j] = tanhf(s);
  }
  __syncthreads();
  if (tid < 64) {                              // actor layer 2
    float s = a2[tid];
    for (int k = 0; k < 64; ++k) s += hA[k] * A2[k * 64 + tid];
    hA2[tid] = tanhf(s);
  } else if (tid < 128) {                      // critic layer 2
    int j = tid - 64; float s = c2[j];
    for (int k = 0; k < 64; ++k) s += hB[k] * C2[k * 64 + j];
    hB2[j] = tanhf(s);
  }
  __syncthreads();
  if (tid < 32) {                              // logits
    float s = a3[tid];
    for (int k = 0; k < 64; ++k) s += hA2[k] * A3[k * 32 + tid];
    lg[tid] = s;
  }
  if (tid == 32) {                             // value
    float s = c3[0];
    for (int k = 0; k < 64; ++k) s += hB2[k] * C3[k];
    valS = s;
  }
  __syncthreads();
  if (tid == 0) {                              // log_softmax / logp / entropy
    float mx = lg[0];
    for (int j = 1; j < 32; ++j) mx = fmaxf(mx, lg[j]);
    float se = 0.0f;
    for (int j = 0; j < 32; ++j) se += __expf(lg[j] - mx);
    float lse = __logf(se) + mx;
    int act = action[b];
    float logp = lg[act] - lse;
    float ent = 0.0f;
    for (int j = 0; j < 32; ++j) { float lp = lg[j] - lse; ent -= __expf(lp) * lp; }
    out[b]              = logp;
    out[BATCHc + b]     = ent;
    out[2 * BATCHc + b] = valS;
  }
}

// ---------------- launch ----------------
extern "C" void kernel_launch(void* const* d_in, const int* in_sizes, int n_in,
                              void* d_out, int out_size, void* d_ws, size_t ws_size,
                              hipStream_t stream) {
  const float* obs    = (const float*)d_in[0];
  const int*   action = (const int*)  d_in[1];
  const float* Wf = (const float*)d_in[2];  const float* bf = (const float*)d_in[3];
  const float* Wi = (const float*)d_in[4];  const float* bi = (const float*)d_in[5];
  const float* Wc = (const float*)d_in[6];  const float* bc = (const float*)d_in[7];
  const float* Wo = (const float*)d_in[8];  const float* bo = (const float*)d_in[9];
  const float* W1 = (const float*)d_in[10]; const float* b1 = (const float*)d_in[11];
  const float* W2 = (const float*)d_in[12]; const float* b2 = (const float*)d_in[13];
  const float* A1 = (const float*)d_in[14]; const float* a1 = (const float*)d_in[15];
  const float* A2 = (const float*)d_in[16]; const float* a2 = (const float*)d_in[17];
  const float* A3 = (const float*)d_in[18]; const float* a3 = (const float*)d_in[19];
  const float* C1 = (const float*)d_in[20]; const float* c1 = (const float*)d_in[21];
  const float* C2 = (const float*)d_in[22]; const float* c2 = (const float*)d_in[23];
  const float* C3 = (const float*)d_in[24]; const float* c3 = (const float*)d_in[25];
  float* out = (float*)d_out;

  char* ws = (char*)d_ws;
  _Float16* obs16 = (_Float16*)ws;
  _Float16* wpack = (_Float16*)(ws + OBS16_BYTES);
  float*    biasg = (float*)   (ws + OBS16_BYTES + WPACK_BYTES);
  float*    hlast = (float*)   (ws + OBS16_BYTES + WPACK_BYTES + BIAS_BYTES);

  // allow >64KB dynamic LDS for the persistent LSTM kernel (idempotent, host-side)
  (void)hipFuncSetAttribute((const void*)k_lstm,
                            hipFuncAttributeMaxDynamicSharedMemorySize,
                            (int)LDS_BYTES);

  // prep
  k_convert_obs<<<(BATCHc * SEQc * OBS_DIMc) / (256 * 4), 256, 0, stream>>>(obs, obs16);
  k_pack_w<<<(KTILES * NTILES * 32 * 16) / 256, 256, 0, stream>>>(Wf, Wi, Wc, Wo, wpack);
  k_pack_bias<<<NGATEc / 256, 256, 0, stream>>>(bf, bi, bc, bo, biasg);

  // persistent LSTM over 512 steps (WMMA f16 gate GEMM, h/c resident in LDS)
  k_lstm<<<BATCHc / 16, 512, LDS_BYTES, stream>>>(obs16, wpack, biasg, hlast);

  // dense layers + actor/critic heads + log_softmax/entropy/value
  k_heads<<<BATCHc, 256, 0, stream>>>(hlast, action, W1, b1, W2, b2,
                                      A1, a1, A2, a2, A3, a3,
                                      C1, c1, C2, c2, C3, c3, out);
}
